// Seq2SeqModel_31954556683002
// MI455X (gfx1250) — matrix operand order repaired and checked by reading, not compiled, not run on hardware
//
#include <hip/hip_runtime.h>
#include <hip/hip_bf16.h>

#define B_     64
#define T_     64
#define EMB_   256
#define HID_   512
#define G4_    2048          // 4*HID
#define VOCAB_ 32000

typedef __bf16 bf16_t;
typedef __attribute__((ext_vector_type(16))) __bf16 v16bf;
typedef __attribute__((ext_vector_type(8)))  float  v8f;
typedef __attribute__((ext_vector_type(4)))  int    v4i;

union BfPack { v16bf v; v4i q[2]; };

__device__ __forceinline__ v16bf load_pair(const bf16_t* p0, const bf16_t* p1) {
  BfPack p;
  p.q[0] = *(const v4i*)p0;
  p.q[1] = *(const v4i*)p1;
  return p.v;
}

// A-matrix 16x32 bf16 layout (ISA 7.12.2): lane<16 holds K[0..7],K[16..23]; lane>=16 K[8..15],K[24..31]
__device__ __forceinline__ v16bf load_A(const bf16_t* row, int k0, int hi) {
  const int ka = k0 + hi * 8;
  return load_pair(row + ka, row + ka + 16);
}
// B-matrix 32x16 bf16 layout: lane<16 holds K[0..7],K[16..23]; lane>=16 K[8..15],K[24..31] (column = lane&15)
__device__ __forceinline__ v16bf load_B(const bf16_t* row, int k0, int hi) {
  const int kb = k0 + hi * 8;
  return load_pair(row + kb, row + kb + 16);
}

__device__ __forceinline__ v8f v8f_zero() {
  v8f z = {0.f, 0.f, 0.f, 0.f, 0.f, 0.f, 0.f, 0.f};
  return z;
}

// ---- CDNA5 async global->LDS copy (ASYNCcnt-tracked), per-lane 16B ----
__device__ __forceinline__ void async_ld_b128(unsigned int ldsaddr, const void* g) {
  asm volatile("global_load_async_to_lds_b128 %0, %1, off"
               :: "v"(ldsaddr), "v"((unsigned long long)(uintptr_t)g)
               : "memory");
}
__device__ __forceinline__ void wait_async0() {
  asm volatile("s_wait_asynccnt 0" ::: "memory");
}

// ---------------- register-pipelined strip GEMM (used by recurrent LSTM part) ----------------
// C[m0+0..16*ROWS-1, n0..n0+15] += A[m,:K] * W[n,:K]^T  (A row-major MxK, W row-major NxK, bf16)
template<int ROWS>
__device__ __forceinline__ void gemm_strip(const bf16_t* __restrict__ A, int lda,
                                           const bf16_t* __restrict__ W, int ldw,
                                           int m0, int n0, int K, int lane,
                                           v8f acc[ROWS]) {
  const int laneM = lane & 15;
  const int hi    = lane >> 4;
  const bf16_t* wrow = W + (size_t)(n0 + laneM) * ldw;
  v16bf bv = load_B(wrow, 0, hi);
  v16bf av[ROWS];
  #pragma unroll
  for (int r = 0; r < ROWS; ++r)
    av[r] = load_A(A + (size_t)(m0 + r * 16 + laneM) * lda, 0, hi);
  #pragma unroll 2
  for (int k0 = 0; k0 < K; k0 += 32) {
    v16bf bn = bv;
    v16bf an[ROWS];
    #pragma unroll
    for (int r = 0; r < ROWS; ++r) an[r] = av[r];
    if (k0 + 32 < K) {   // prefetch next k-step while this one computes
      bn = load_B(wrow, k0 + 32, hi);
      #pragma unroll
      for (int r = 0; r < ROWS; ++r)
        an[r] = load_A(A + (size_t)(m0 + r * 16 + laneM) * lda, k0 + 32, hi);
    }
    #pragma unroll
    for (int r = 0; r < ROWS; ++r)
      acc[r] = __builtin_amdgcn_wmma_f32_16x16x32_bf16(
          false, av[r], false, bv, (short)0, acc[r], false, false);
    bv = bn;
    #pragma unroll
    for (int r = 0; r < ROWS; ++r) av[r] = an[r];
  }
}

// ---------------- block-cooperative GEMM with async-LDS A staging ----------------
// Block = 256 threads (8 waves). All waves share M-strip [m0, m0+64); wave w owns N-tile n0(w).
// A chunks (64 rows x 32 K, 4KB) double-buffered in LDS via global_load_async_to_lds_b128;
// each thread stages 16B. B tiles prefetched one k-step ahead into registers.
template<int K>
__device__ __forceinline__ void gemm_lds(const bf16_t* __restrict__ A, int lda,
                                         const bf16_t* __restrict__ W, int ldw,
                                         int m0, int n0, bf16_t (*sA)[64 * 32],
                                         v8f acc[4]) {
  const int lane  = threadIdx.x & 31;
  const int laneM = lane & 15;
  const int hi    = lane >> 4;
  const int sidx  = threadIdx.x * 16;            // byte index within 4KB chunk
  const int srow  = sidx >> 6;                   // staged row (64B per row)
  const int scol  = sidx & 63;                   // byte within row
  const char* grow = (const char*)(A + (size_t)(m0 + srow) * lda) + scol;
  const unsigned int lds0 = (unsigned int)(uintptr_t)&sA[0][0] + (unsigned int)sidx;
  const unsigned int lds1 = (unsigned int)(uintptr_t)&sA[1][0] + (unsigned int)sidx;
  constexpr int NC = K / 32;

  async_ld_b128(lds0, grow);                     // prologue: chunk 0
  const bf16_t* wrow = W + (size_t)(n0 + laneM) * ldw;
  v16bf bv = load_B(wrow, 0, hi);
  wait_async0();
  __syncthreads();

  #pragma unroll 2
  for (int kc = 0; kc < NC; ++kc) {
    const int cur = kc & 1;
    v16bf bn = bv;
    if (kc + 1 < NC) {
      async_ld_b128(cur ? lds0 : lds1, grow + (kc + 1) * 64);  // overlap next A chunk
      bn = load_B(wrow, (kc + 1) * 32, hi);                    // prefetch next B tile
    }
    const bf16_t* s = &sA[cur][0] + (size_t)laneM * 32 + hi * 8;
    #pragma unroll
    for (int r = 0; r < 4; ++r) {
      v16bf av = load_pair(s + r * 512, s + r * 512 + 16);     // ds_load_b128 x2
      acc[r] = __builtin_amdgcn_wmma_f32_16x16x32_bf16(
          false, av, false, bv, (short)0, acc[r], false, false);
    }
    bv = bn;
    wait_async0();
    __syncthreads();
  }
}

// ---------------- utility kernels ----------------

__global__ void cvt_kernel(const float* __restrict__ s, bf16_t* __restrict__ d, int n) {
  for (int i = blockIdx.x * blockDim.x + threadIdx.x; i < n; i += gridDim.x * blockDim.x)
    d[i] = (bf16_t)s[i];
}

__global__ void bias_kernel(const float* __restrict__ a, const float* __restrict__ b,
                            float* __restrict__ o, int n) {
  int i = blockIdx.x * blockDim.x + threadIdx.x;
  if (i < n) o[i] = a[i] + b[i];
}

// Build encoder/decoder input embeddings, bf16, layout [t*B + b, EMB]
__global__ void embed_kernel(const int* __restrict__ toks, const float* __restrict__ emb,
                             bf16_t* __restrict__ Xenc, bf16_t* __restrict__ Xdec) {
  const int tb = blockIdx.x;       // t*B + b
  const int t  = tb >> 6;
  const int b  = tb & 63;
  const int e  = threadIdx.x;      // 0..255
  const int tokE = toks[b * T_ + t];
  const int tokD = (t == 0) ? 1 : toks[b * T_ + t - 1];   // INIT token = 1, teacher forcing
  Xenc[(size_t)tb * EMB_ + e] = (bf16_t)emb[(size_t)tokE * EMB_ + e];
  Xdec[(size_t)tb * EMB_ + e] = (bf16_t)emb[(size_t)tokD * EMB_ + e];
}

__global__ void len_kernel(const int* __restrict__ toks, int* __restrict__ len) {
  int b = threadIdx.x;
  if (b < B_) {
    int n = 0;
    for (int t = 0; t < T_; ++t) n += (toks[b * T_ + t] != 0);
    len[b] = n;
  }
}

// ---------------- batched input GEMM: C[4096,2048] = X[4096,256] @ Wih^T + (bih+bhh) ----------------
// Block covers 64 M-rows x 128 N-cols. Grid = 64 strips * 16 N-groups = 1024 blocks.

__global__ void xw_gemm_kernel(const bf16_t* __restrict__ X, const bf16_t* __restrict__ Wih,
                               const float* __restrict__ bias, float* __restrict__ C) {
  __shared__ bf16_t sA[2][64 * 32];
  const int wv = threadIdx.x >> 5;
  const int NG = G4_ / 128;              // 16
  const int m0 = (blockIdx.x / NG) * 64;
  const int n0 = (blockIdx.x % NG) * 128 + wv * 16;
  v8f acc[4];
  #pragma unroll
  for (int r = 0; r < 4; ++r) acc[r] = v8f_zero();
  gemm_lds<EMB_>(X, EMB_, Wih, EMB_, m0, n0, sA, acc);
  const int lane = threadIdx.x & 31;
  const int hi = lane >> 4, ln = lane & 15;
  const float bv = bias[n0 + ln];
  #pragma unroll
  for (int r = 0; r < 4; ++r)
    #pragma unroll
    for (int g = 0; g < 8; ++g)
      C[(size_t)(m0 + r * 16 + g + 8 * hi) * G4_ + n0 + ln] = acc[r][g] + bv;
}

// ---------------- persistent LSTM (16 blocks x 256 threads = 128 waves) ----------------

__device__ __forceinline__ void gsync(int* bar, int& round) {
  __syncthreads();
  if (threadIdx.x == 0) {
    __threadfence();
    atomicAdd(bar, 1);
    ++round;
    const int target = round * (int)gridDim.x;
    while (__atomic_load_n((volatile int*)bar, __ATOMIC_RELAXED) < target)
      __builtin_amdgcn_s_sleep(1);
    __threadfence();
  }
  __syncthreads();
}

// MODE 0: encoder (masked by len, h/c start at 0). MODE 1: decoder (h kept from encoder, c=0, emits hs bf16)
template<int MODE>
__global__ void lstm_kernel(const float* __restrict__ XW,       // [T*B, 2048] precomputed x@Wih^T + biases
                            const bf16_t* __restrict__ Whh,     // [2048, 512] bf16
                            float* __restrict__ h, float* __restrict__ c,
                            bf16_t* __restrict__ hbf,
                            float* __restrict__ gates,          // [64, 2048] scratch
                            const int* __restrict__ len,
                            bf16_t* __restrict__ hs,            // [T*B, 512] (decoder only)
                            int* __restrict__ bar) {
  const int tid   = blockIdx.x * blockDim.x + threadIdx.x;   // 0..4095
  const int nthr  = gridDim.x * blockDim.x;                  // 4096
  const int wave  = tid >> 5;                                // 0..127 == N-tile id
  const int lane  = threadIdx.x & 31;
  int round = 0;

  for (int idx = tid; idx < B_ * HID_; idx += nthr) {
    float hv = (MODE == 0) ? 0.f : h[idx];
    if (MODE == 0) h[idx] = 0.f;
    c[idx]   = 0.f;
    hbf[idx] = (bf16_t)hv;
  }
  gsync(bar, round);

  const int hi = lane >> 4, ln = lane & 15;
  const int n0 = wave * 16;

  for (int t = 0; t < T_; ++t) {
    // gates[64,2048] = XW[t] + hbf[64,512] @ Whh^T ; each wave: 1 N-tile x 4 M-tiles
    {
      v8f acc[4];
      #pragma unroll
      for (int r = 0; r < 4; ++r)
        #pragma unroll
        for (int g = 0; g < 8; ++g)
          acc[r][g] = XW[(size_t)(t * B_ + r * 16 + g + 8 * hi) * G4_ + n0 + ln];
      gemm_strip<4>(hbf, HID_, Whh, HID_, 0, n0, HID_, lane, acc);
      #pragma unroll
      for (int r = 0; r < 4; ++r)
        #pragma unroll
        for (int g = 0; g < 8; ++g)
          gates[(size_t)(r * 16 + g + 8 * hi) * G4_ + n0 + ln] = acc[r][g];
    }
    gsync(bar, round);

    // elementwise cell update (PyTorch gate order i,f,g,o)
    for (int idx = tid; idx < B_ * HID_; idx += nthr) {
      const int b = idx >> 9;
      const int j = idx & (HID_ - 1);
      const float* gr = gates + (size_t)b * G4_;
      const float ig = gr[j], fg = gr[HID_ + j], gg = gr[2 * HID_ + j], og = gr[3 * HID_ + j];
      const float si = 1.f / (1.f + __expf(-ig));
      const float sf = 1.f / (1.f + __expf(-fg));
      const float so = 1.f / (1.f + __expf(-og));
      float cn = sf * c[idx] + si * tanhf(gg);
      float hn = so * tanhf(cn);
      if (MODE == 0) {
        const bool act = (t < len[b]);
        cn = act ? cn : c[idx];
        hn = act ? hn : h[idx];
      }
      c[idx] = cn;
      h[idx] = hn;
      hbf[idx] = (bf16_t)hn;
      if (MODE == 1) hs[(size_t)(t * B_ + b) * HID_ + j] = (bf16_t)hn;
    }
    gsync(bar, round);
  }
}

// ---------------- output projection: logits[b,t,v] = hs[t*B+b,:] . linW[v,:] + lin_b[v] ----------------
// Block covers 64 M-rows x 128 vocab cols. Grid = 64 strips * 250 N-groups = 16000 blocks.

__global__ void proj_kernel(const bf16_t* __restrict__ hs, const bf16_t* __restrict__ Wl,
                            const float* __restrict__ lb, float* __restrict__ out) {
  __shared__ bf16_t sA[2][64 * 32];
  const int wv = threadIdx.x >> 5;
  const int NG = VOCAB_ / 128;           // 250
  const int m0 = (blockIdx.x / NG) * 64;
  const int n0 = (blockIdx.x % NG) * 128 + wv * 16;
  v8f acc[4];
  #pragma unroll
  for (int r = 0; r < 4; ++r) acc[r] = v8f_zero();
  gemm_lds<HID_>(hs, HID_, Wl, HID_, m0, n0, sA, acc);
  const int lane = threadIdx.x & 31;
  const int hi = lane >> 4, ln = lane & 15;
  const float bv = lb[n0 + ln];
  #pragma unroll
  for (int r = 0; r < 4; ++r)
    #pragma unroll
    for (int g = 0; g < 8; ++g) {
      const int m = m0 + r * 16 + g + 8 * hi;   // m = t*B + b
      const int t = m >> 6, b = m & 63;
      out[((size_t)b * T_ + t) * VOCAB_ + n0 + ln] = acc[r][g] + bv;
    }
}

// ---------------- host ----------------

extern "C" void kernel_launch(void* const* d_in, const int* in_sizes, int n_in,
                              void* d_out, int out_size, void* d_ws, size_t ws_size,
                              hipStream_t stream) {
  (void)in_sizes; (void)n_in; (void)out_size; (void)ws_size;
  const int*   toks = (const int*)  d_in[0];
  const float* emb  = (const float*)d_in[1];
  const float* eWih = (const float*)d_in[2];
  const float* eWhh = (const float*)d_in[3];
  const float* ebih = (const float*)d_in[4];
  const float* ebhh = (const float*)d_in[5];
  const float* dWih = (const float*)d_in[6];
  const float* dWhh = (const float*)d_in[7];
  const float* dbih = (const float*)d_in[8];
  const float* dbhh = (const float*)d_in[9];
  const float* linW = (const float*)d_in[10];
  const float* linb = (const float*)d_in[11];
  float* out = (float*)d_out;

  char* ws = (char*)d_ws;
  size_t off = 0;
  auto take = [&](size_t elems, size_t esz) -> char* {
    char* p = ws + off;
    off += (elems * esz + 255) & ~(size_t)255;
    return p;
  };
  bf16_t* eWih_bf = (bf16_t*)take((size_t)G4_ * EMB_, 2);
  bf16_t* eWhh_bf = (bf16_t*)take((size_t)G4_ * HID_, 2);
  bf16_t* dWih_bf = (bf16_t*)take((size_t)G4_ * EMB_, 2);
  bf16_t* dWhh_bf = (bf16_t*)take((size_t)G4_ * HID_, 2);
  bf16_t* linW_bf = (bf16_t*)take((size_t)VOCAB_ * HID_, 2);
  bf16_t* Xenc    = (bf16_t*)take((size_t)T_ * B_ * EMB_, 2);
  bf16_t* Xdec    = (bf16_t*)take((size_t)T_ * B_ * EMB_, 2);
  float*  XWenc   = (float*) take((size_t)T_ * B_ * G4_, 4);
  float*  XWdec   = (float*) take((size_t)T_ * B_ * G4_, 4);
  float*  gates   = (float*) take((size_t)B_ * G4_, 4);
  float*  hbuf    = (float*) take((size_t)B_ * HID_, 4);
  float*  cbuf    = (float*) take((size_t)B_ * HID_, 4);
  bf16_t* hbf     = (bf16_t*)take((size_t)B_ * HID_, 2);
  bf16_t* hs_bf   = (bf16_t*)take((size_t)T_ * B_ * HID_, 2);
  float*  ebias   = (float*) take(G4_, 4);
  float*  dbias   = (float*) take(G4_, 4);
  int*    len     = (int*)   take(64, 4);
  int*    bar     = (int*)   take(64, 4);

  // precision conversion + small prep
  cvt_kernel<<<512, 256, 0, stream>>>(eWih, eWih_bf, G4_ * EMB_);
  cvt_kernel<<<512, 256, 0, stream>>>(eWhh, eWhh_bf, G4_ * HID_);
  cvt_kernel<<<512, 256, 0, stream>>>(dWih, dWih_bf, G4_ * EMB_);
  cvt_kernel<<<512, 256, 0, stream>>>(dWhh, dWhh_bf, G4_ * HID_);
  cvt_kernel<<<4096, 256, 0, stream>>>(linW, linW_bf, VOCAB_ * HID_);
  bias_kernel<<<(G4_ + 255) / 256, 256, 0, stream>>>(ebih, ebhh, ebias, G4_);
  bias_kernel<<<(G4_ + 255) / 256, 256, 0, stream>>>(dbih, dbhh, dbias, G4_);
  embed_kernel<<<T_ * B_, EMB_, 0, stream>>>(toks, emb, Xenc, Xdec);
  len_kernel<<<1, 64, 0, stream>>>(toks, len);

  // hoisted input-side GEMMs: [4096,256] x [256,2048] + bias
  xw_gemm_kernel<<<1024, 256, 0, stream>>>(Xenc, eWih_bf, ebias, XWenc);
  xw_gemm_kernel<<<1024, 256, 0, stream>>>(Xdec, dWih_bf, dbias, XWdec);

  // recurrent parts: persistent 16-block kernels with grid barrier (counter must start at 0 each call)
  hipMemsetAsync(bar, 0, 256, stream);
  lstm_kernel<0><<<16, 256, 0, stream>>>(XWenc, eWhh_bf, hbuf, cbuf, hbf, gates, len,
                                         (bf16_t*)nullptr, bar);
  hipMemsetAsync(bar, 0, 256, stream);
  lstm_kernel<1><<<16, 256, 0, stream>>>(XWdec, dWhh_bf, hbuf, cbuf, hbf, gates, len,
                                         hs_bf, bar);

  // projection: [4096,512] x [512,32000] + lin_b
  proj_kernel<<<16000, 256, 0, stream>>>(hs_bf, linW_bf, linb, out);
}
